// GraphConvolution_38405597561663
// MI455X (gfx1250) — compile-verified, matrix operand order checked by reading, not run on hardware
//
#include <hip/hip_runtime.h>
#include <hip/hip_bf16.h>

#define NNODES 50000   // divisible by 16 -> no partial WMMA tiles
#define NEDGES 640000
#define DF     128

typedef __attribute__((ext_vector_type(16))) __bf16 v16bf;
typedef __attribute__((ext_vector_type(8)))  float  v8f;

__device__ __forceinline__ __bf16 f2bf(float f) {
    unsigned u = __builtin_bit_cast(unsigned, f);
    u += 0x7FFFu + ((u >> 16) & 1u);            // round-to-nearest-even
    unsigned short h = (unsigned short)(u >> 16);
    return __builtin_bit_cast(__bf16, h);
}

// Native CDNA5 no-return float atomic at device scope (L2 = coherence point).
__device__ __forceinline__ void atomic_add_f32_dev(float* addr, float v) {
    asm volatile("global_atomic_add_f32 %0, %1, off scope:SCOPE_DEV"
                 :: "v"(addr), "v"(v) : "memory");
}

// ---------------------------------------------------------------------------
// Kernel 0: build WT[n][k] = bf16( W[k][n] ), n in [0,256) over [w1|w2]
// ---------------------------------------------------------------------------
__global__ __launch_bounds__(256) void build_wt_kernel(
    const float* __restrict__ w1, const float* __restrict__ w2,
    __bf16* __restrict__ WT)
{
    int idx = blockIdx.x * 256 + threadIdx.x;   // 0..32767 == n*128 + k
    int n = idx >> 7;
    int k = idx & 127;
    float v = (n < DF) ? w1[k * DF + n] : w2[k * DF + (n - DF)];
    WT[idx] = f2bf(v);
}

// ---------------------------------------------------------------------------
// Kernel 1: WMMA GEMM. Each wave: 16 rows x all 256 cols (16 tiles x 4 WMMA).
//   cols [0,128)   -> d_out  = X@w1 + bias
//   cols [128,256) -> X1(ws) = X@w2
// Block = 256 threads = 8 waves = 128 rows per block.
// ---------------------------------------------------------------------------
__global__ __launch_bounds__(256) void gemm_wmma_kernel(
    const float* __restrict__ X, const __bf16* __restrict__ WT,
    const float* __restrict__ bias,
    float* __restrict__ out, float* __restrict__ X1)
{
    const int lane = threadIdx.x & 31;
    const int wave = threadIdx.x >> 5;
    const int rowBase = blockIdx.x * 128 + wave * 16;
    if (rowBase >= NNODES) return;              // uniform per wave: EXEC stays all-1

    const int m    = lane & 15;
    const int half = lane >> 4;

    // Load A once: 16x128 f32 -> four 16x32 bf16 fragments (ISA §7.12.2 layout).
    // NNODES % 16 == 0, so every row of an active tile is in-bounds.
    v16bf a[4];
    {
        const float* xr = X + (size_t)(rowBase + m) * DF;
#pragma unroll
        for (int kc = 0; kc < 4; ++kc) {
            const int kb = kc * 32;
            const float4 lo0 = *(const float4*)(xr + kb + half * 8);
            const float4 lo1 = *(const float4*)(xr + kb + half * 8 + 4);
            const float4 hi0 = *(const float4*)(xr + kb + 16 + half * 8);
            const float4 hi1 = *(const float4*)(xr + kb + 16 + half * 8 + 4);
            a[kc][0]  = f2bf(lo0.x); a[kc][1]  = f2bf(lo0.y);
            a[kc][2]  = f2bf(lo0.z); a[kc][3]  = f2bf(lo0.w);
            a[kc][4]  = f2bf(lo1.x); a[kc][5]  = f2bf(lo1.y);
            a[kc][6]  = f2bf(lo1.z); a[kc][7]  = f2bf(lo1.w);
            a[kc][8]  = f2bf(hi0.x); a[kc][9]  = f2bf(hi0.y);
            a[kc][10] = f2bf(hi0.z); a[kc][11] = f2bf(hi0.w);
            a[kc][12] = f2bf(hi1.x); a[kc][13] = f2bf(hi1.y);
            a[kc][14] = f2bf(hi1.z); a[kc][15] = f2bf(hi1.w);
        }
    }

#pragma unroll 1
    for (int t = 0; t < 16; ++t) {
        const int n = t * 16 + m;               // 0..255, lane's output column
        const __bf16* wt = WT + (size_t)n * DF;

        // Load ALL four B fragments first -> one 8x b128 load clause,
        // lets the scheduler overlap loads with the WMMA chain.
        v16bf bm[4];
#pragma unroll
        for (int kc = 0; kc < 4; ++kc) {
            const int kb = kc * 32;
#pragma unroll
            for (int j = 0; j < 8; ++j) {
                bm[kc][j]     = wt[kb + half * 8 + j];
                bm[kc][8 + j] = wt[kb + 16 + half * 8 + j];
            }
        }

        v8f c = {};
#pragma unroll
        for (int kc = 0; kc < 4; ++kc) {
            c = __builtin_amdgcn_wmma_f32_16x16x32_bf16(
                    false, a[kc], false, bm[kc], (short)0, c, false, false);
        }

        // Epilogue: lanes 0-15 -> rows M=0..7, lanes 16-31 -> rows M=8..15.
        // No guards needed: tiles are always full.
        const int mOff = half * 8;
        if (n < DF) {
            const float bv = bias[n];
            float* o = out + (size_t)(rowBase + mOff) * DF + n;
#pragma unroll
            for (int r = 0; r < 8; ++r)
                o[(size_t)r * DF] = c[r] + bv;
        } else {
            float* o = X1 + (size_t)(rowBase + mOff) * DF + (n - DF);
#pragma unroll
            for (int r = 0; r < 8; ++r)
                o[(size_t)r * DF] = c[r];
        }
    }
}

// ---------------------------------------------------------------------------
// Kernel 2: edge scatter. One wave per edge; lane handles a float4 chunk.
//   out[dst] += w * X1[src]   (82M native global_atomic_add_f32, L2-resident)
// ---------------------------------------------------------------------------
__global__ __launch_bounds__(256) void edge_scatter_kernel(
    const float* __restrict__ X1, const float* __restrict__ ew,
    const int* __restrict__ es, const int* __restrict__ ed,
    float* __restrict__ out)
{
    const unsigned idx = blockIdx.x * 256u + threadIdx.x;
    const unsigned e = idx >> 5;                // edge id (uniform per wave)
    const unsigned c = (idx & 31u) * 4u;        // feature chunk
    const int   s = es[e];
    const int   d = ed[e];
    const float w = ew[e];
    const float4 v = *(const float4*)(X1 + (size_t)s * DF + c);
    float* o = out + (size_t)d * DF + c;
    atomic_add_f32_dev(o + 0, w * v.x);
    atomic_add_f32_dev(o + 1, w * v.y);
    atomic_add_f32_dev(o + 2, w * v.z);
    atomic_add_f32_dev(o + 3, w * v.w);
}

// ---------------------------------------------------------------------------
// Kernel 3: in-place ReLU on d_out (float4 vectorized)
// ---------------------------------------------------------------------------
__global__ __launch_bounds__(256) void relu_kernel(float4* __restrict__ out)
{
    const int i = blockIdx.x * 256 + threadIdx.x;
    float4 v = out[i];
    v.x = fmaxf(v.x, 0.0f);
    v.y = fmaxf(v.y, 0.0f);
    v.z = fmaxf(v.z, 0.0f);
    v.w = fmaxf(v.w, 0.0f);
    out[i] = v;
}

// ---------------------------------------------------------------------------
extern "C" void kernel_launch(void* const* d_in, const int* in_sizes, int n_in,
                              void* d_out, int out_size, void* d_ws, size_t ws_size,
                              hipStream_t stream) {
    const float* X  = (const float*)d_in[0];
    const float* w1 = (const float*)d_in[1];
    const float* w2 = (const float*)d_in[2];
    const float* b  = (const float*)d_in[3];
    const float* ew = (const float*)d_in[4];
    const int*   es = (const int*)d_in[5];
    const int*   ed = (const int*)d_in[6];
    float* out = (float*)d_out;

    // workspace: [0,64KB) WT bf16 [256][128]; [64KB, ...) X1 f32 [50000][128]
    __bf16* WT = (__bf16*)d_ws;
    float*  X1 = (float*)((char*)d_ws + 65536);

    build_wt_kernel<<<(256 * DF) / 256, 256, 0, stream>>>(w1, w2, WT);

    gemm_wmma_kernel<<<(NNODES + 127) / 128, 256, 0, stream>>>(X, WT, b, out, X1);

    edge_scatter_kernel<<<(NEDGES * 32) / 256, 256, 0, stream>>>(X1, ew, es, ed, out);

    relu_kernel<<<(NNODES * DF / 4) / 256, 256, 0, stream>>>((float4*)out);
}
